// SemanticMaskVectorQuantizer_75342316307159
// MI455X (gfx1250) — matrix operand order repaired and checked by reading, not compiled, not run on hardware
//
#include <hip/hip_runtime.h>

// ---------------------------------------------------------------------------
// VQ-VAE vector quantizer for MI455X (gfx1250, wave32, WMMA).
// z_e: [16,256,32,32] f32, embed: [8192,256] f32.
// Fused GEMM(+argmin) via v_wmma_f32_16x16x32_bf16, fp32 everywhere else.
// Round 4: 32-code LDS tiles (16 WMMA/iter, half the barriers), enorm slice
// cached in LDS (kills the exposed per-tile global load + loadcnt wait),
// software-pipelined B reads (one ds_load pair in flight behind each WMMA),
// async DMA staging (ASYNCcnt), double-buffered LDS, 4-way split-K.
// ---------------------------------------------------------------------------

typedef __attribute__((ext_vector_type(16))) __bf16 v16bf;
typedef __attribute__((ext_vector_type(8)))  float  v8f;

#define D_DIM    256
#define K_CODES  8192
#define N_TOT    16384
#define HW       1024
#define CHW      262144          // 256*1024
#define ROWS_PB  128             // rows of z per block (8 waves * 16)
#define BT_STRIDE 264            // padded LDS row stride (bf16 elems) -> no bank conflicts
#define SPLITK   4
#define KSLICE   (K_CODES / SPLITK)        // 2048 codes per split
#define TILE_C   32                        // codes per LDS tile
#define CT_PER   (KSLICE / TILE_C)         // 64 tiles per split

// output layout (float elements)
#define OUT_ZQST 0
#define OUT_ZQ   4194304
#define OUT_CODE 8388608
#define OUT_SCAL 8404992

__device__ __forceinline__ unsigned short f2bf(float f) {
  unsigned int u = __float_as_uint(f);
  u += 0x7FFFu + ((u >> 16) & 1u);            // round-to-nearest-even
  return (unsigned short)(u >> 16);
}

// async DMA: 64 bytes global -> LDS per lane (4x b128; offset applies to both
// the LDS and global side per ISA pseudocode). Tracked by ASYNCcnt.
__device__ __forceinline__ void async_copy64(unsigned lds_off, const void* gptr) {
  asm volatile(
      "global_load_async_to_lds_b128 %0, %1, off\n\t"
      "global_load_async_to_lds_b128 %0, %1, off offset:16\n\t"
      "global_load_async_to_lds_b128 %0, %1, off offset:32\n\t"
      "global_load_async_to_lds_b128 %0, %1, off offset:48"
      :: "v"(lds_off), "v"((unsigned long long)(uintptr_t)gptr)
      : "memory");
}
__device__ __forceinline__ void async_wait0() {
  asm volatile("s_wait_asynccnt 0x0" ::: "memory");
}

// ---------------------------------------------------------------- zero counts
__global__ void vq_zero(int* __restrict__ cnt) {
  int i = blockIdx.x * blockDim.x + threadIdx.x;
  if (i < K_CODES) cnt[i] = 0;
}

// ------------------------------------------------- codebook: f32->bf16 + norms
__global__ void vq_prep_embed(const float* __restrict__ w,
                              unsigned short* __restrict__ ebf,
                              float* __restrict__ enorm) {
  __shared__ float red[256];
  const int k = blockIdx.x;
  const int t = threadIdx.x;
  float x = w[k * D_DIM + t];
  ebf[k * D_DIM + t] = f2bf(x);
  red[t] = x * x;
  __syncthreads();
  for (int s = 128; s > 0; s >>= 1) { if (t < s) red[t] += red[t + s]; __syncthreads(); }
  if (t == 0) enorm[k] = red[0];
}

// ------------------------------------ z: NCHW f32 -> [N,D] bf16 (N=b*HW+hw)
__global__ void vq_prep_z(const float* __restrict__ z,
                          unsigned short* __restrict__ zbf) {
  int id = blockIdx.x * 256 + threadIdx.x;   // < N_TOT*D_DIM
  int n = id >> 8, c = id & 255;
  int b = n >> 10, hw = n & 1023;
  zbf[id] = f2bf(z[(size_t)b * CHW + (size_t)c * HW + hw]);
}

// --------------------------- main: GEMM + argmin over one codebook split
__global__ __launch_bounds__(256) void vq_main(
    const unsigned short* __restrict__ zbf, // [N,D] bf16
    const unsigned short* __restrict__ ebf, // [K,D] bf16
    const float* __restrict__ enorm,        // [K] ||e||^2
    float* __restrict__ wsDist,             // [SPLITK][N] candidate dist
    int*   __restrict__ wsIdx) {            // [SPLITK][N] candidate code
  __shared__ unsigned short Bt[2][TILE_C * BT_STRIDE]; // double-buffered tiles
  __shared__ float enLds[KSLICE];                      // split's ||e||^2 slice

  const int tid  = threadIdx.x;
  const int wave = tid >> 5;
  const int lane = tid & 31;
  const int half = lane >> 4;
  const int l16  = lane & 15;
  const int n0   = blockIdx.x * ROWS_PB;
  const int kbase = blockIdx.y * KSLICE;
  const int myrow = n0 + wave * 16 + l16;

  // --- preload A (16x256 bf16 per wave), ISA lane layout:
  // lanes 0-15: K = j*32+[0..7],[16..23]; lanes 16-31: +8 shift.
  v16bf a[8];
  {
    const unsigned short* zr = zbf + (size_t)myrow * D_DIM;
#pragma unroll
    for (int j = 0; j < 8; j++) {
      uint4* ap = reinterpret_cast<uint4*>(&a[j]);
      ap[0] = *(const uint4*)(zr + j * 32 + 8 * half);
      ap[1] = *(const uint4*)(zr + j * 32 + 16 + 8 * half);
    }
  }

  float best[8];
  int   bidx[8];
#pragma unroll
  for (int r = 0; r < 8; r++) { best[r] = 3.4e38f; bidx[r] = 0; }

  // DMA mapping: 256 threads cover a 32x256 bf16 tile, 64B per thread
  const int loadRow = tid >> 3;              // code row in tile (0..31)
  const int loadCol = (tid & 7) * 32;        // bf16 element offset (64B)
  const unsigned short* eb =
      ebf + (size_t)(kbase + loadRow) * D_DIM + loadCol;
  const unsigned ldsDst0 =
      (unsigned)(uintptr_t)&Bt[0][loadRow * BT_STRIDE + loadCol];
  const unsigned ldsDst1 =
      (unsigned)(uintptr_t)&Bt[1][loadRow * BT_STRIDE + loadCol];

  // prologue: DMA tile 0, stage this split's norms into LDS
  async_copy64(ldsDst0, eb);
#pragma unroll
  for (int i = 0; i < KSLICE / 256; i++)
    enLds[tid + 256 * i] = enorm[kbase + tid + 256 * i];
  async_wait0();
  __syncthreads();

  for (int ct = 0; ct < CT_PER; ct++) {
    const int code0 = kbase + ct * TILE_C;

    // DMA next tile into the other buffer NOW; covered by the WMMA chain.
    // Unconditional: one-past-the-end reads land in adjacent ws (unused).
    async_copy64((ct & 1) ? ldsDst0 : ldsDst1,
                 eb + (size_t)(ct + 1) * TILE_C * D_DIM);

    // two 16-code sub-tiles; software-pipelined B reads (1 pair in flight)
    const unsigned short* bpA = &Bt[ct & 1][l16 * BT_STRIDE + 8 * half];
    const unsigned short* bpB = bpA + 16 * BT_STRIDE;
    v8f acc0 = {0.f, 0.f, 0.f, 0.f, 0.f, 0.f, 0.f, 0.f};
    v8f acc1 = {0.f, 0.f, 0.f, 0.f, 0.f, 0.f, 0.f, 0.f};
    v16bf bpp[2];
#define LOADB(slot, jj)                                                       \
    {                                                                         \
      const unsigned short* s_ =                                              \
          ((jj) < 8) ? (bpA + (jj) * 32) : (bpB + ((jj) - 8) * 32);           \
      uint4* q_ = reinterpret_cast<uint4*>(&bpp[slot]);                       \
      q_[0] = *(const uint4*)(s_);                                            \
      q_[1] = *(const uint4*)(s_ + 16);                                       \
    }
    LOADB(0, 0);
#pragma unroll
    for (int j = 0; j < 16; j++) {
      if (j + 1 < 16) LOADB((j + 1) & 1, j + 1);
      if (j < 8)
        acc0 = __builtin_amdgcn_wmma_f32_16x16x32_bf16(
            false, a[j], false, bpp[j & 1], (short)0, acc0, false, false);
      else
        acc1 = __builtin_amdgcn_wmma_f32_16x16x32_bf16(
            false, a[j - 8], false, bpp[j & 1], (short)0, acc1, false, false);
    }
#undef LOADB

    // dist = ||e||^2 - 2*dot (||z||^2 is a per-row constant); norms from LDS
    const int c0 = code0 + l16;
    const int c1 = c0 + 16;
    const float en0 = enLds[ct * TILE_C + l16];
    const float en1 = enLds[ct * TILE_C + 16 + l16];
#pragma unroll
    for (int r = 0; r < 8; r++) {
      float d0 = en0 - 2.0f * acc0[r];
      if (d0 < best[r]) { best[r] = d0; bidx[r] = c0; }
      float d1 = en1 - 2.0f * acc1[r];
      if (d1 < best[r]) { best[r] = d1; bidx[r] = c1; }
    }

    async_wait0();     // retire next-tile DMA (latency already hidden)
    __syncthreads();   // single barrier per iteration (double-buffered)
  }

  // ---- reduce argmin across the 16 lanes sharing each row-set (deterministic)
#pragma unroll
  for (int off = 8; off >= 1; off >>= 1) {
#pragma unroll
    for (int r = 0; r < 8; r++) {
      float od = __shfl_xor(best[r], off, 32);
      int   oi = __shfl_xor(bidx[r], off, 32);
      if (od < best[r] || (od == best[r] && oi < bidx[r])) { best[r] = od; bidx[r] = oi; }
    }
  }
  // lanes 0-15 hold rows M=0..7 (VGPR r -> M=r); lanes 16-31 hold M=8..15
  if (l16 == 0) {
    const size_t o = (size_t)blockIdx.y * N_TOT + n0 + wave * 16 + half * 8;
#pragma unroll
    for (int r = 0; r < 8; r++) { wsDist[o + r] = best[r]; wsIdx[o + r] = bidx[r]; }
  }
}

// ---------------- combine splits, codes, histogram, gather z_q, losses
__global__ __launch_bounds__(256) void vq_combine(
    const float* __restrict__ z,            // original z_e (NCHW f32)
    const float* __restrict__ w,            // original codebook f32
    const float* __restrict__ wsDist,
    const int*   __restrict__ wsIdx,
    float* __restrict__ out,
    int*   __restrict__ counts,
    float* __restrict__ partial) {
  __shared__ int   finalIdx[ROWS_PB];
  __shared__ float red[256];

  const int tid = threadIdx.x;
  const int n0  = blockIdx.x * ROWS_PB;

  if (tid < ROWS_PB) {
    const int row = n0 + tid;
    float bd = wsDist[row];
    int   bi = wsIdx[row];
#pragma unroll
    for (int s = 1; s < SPLITK; s++) {      // ascending split => first-min ties
      float d = wsDist[(size_t)s * N_TOT + row];
      int   i = wsIdx[(size_t)s * N_TOT + row];
      if (d < bd || (d == bd && i < bi)) { bd = d; bi = i; }
    }
    finalIdx[tid] = bi;
    out[OUT_CODE + row] = (float)bi;
    atomicAdd(&counts[bi], 1);              // int atomics: order-independent
  }
  __syncthreads();

  // gather z_q (f32 codebook), write z_q_st & z_q (NCHW), accumulate MSE
  const int row = tid & 127;
  const int cg  = tid >> 7;
  const int n   = n0 + row;
  const int b   = n >> 10;
  const int hw  = n & 1023;
  const int code = finalIdx[row];
  const float* wr = w + (size_t)code * D_DIM;
  const size_t obase = (size_t)b * CHW + hw;
  float lsum = 0.f;
#pragma unroll 4
  for (int i = 0; i < 128; i++) {
    int c = cg * 128 + i;
    float e = wr[c];
    size_t oi = obase + (size_t)c * HW;
    out[OUT_ZQST + oi] = e;                 // z_q_st == z_q numerically
    out[OUT_ZQ   + oi] = e;
    float d = e - z[oi];
    lsum += d * d;
  }
  red[tid] = lsum;
  __syncthreads();
  for (int s = 128; s > 0; s >>= 1) { if (tid < s) red[tid] += red[tid + s]; __syncthreads(); }
  if (tid == 0) partial[blockIdx.x] = red[0];
}

// ------------------------------------------------ finalize scalars (1 block)
__global__ void vq_final(const float* __restrict__ partial,
                         const int* __restrict__ counts,
                         float* __restrict__ out) {
  __shared__ float fred[256];
  __shared__ int   ired[256];
  __shared__ int   ired2[256];
  const int t = threadIdx.x;

  // loss = sum(partials)/numel  (fixed-tree reduce -> deterministic)
  fred[t] = (t < 128) ? partial[t] : 0.f;
  __syncthreads();
  for (int k = 128; k > 0; k >>= 1) { if (t < k) fred[t] += fred[t + k]; __syncthreads(); }
  const float loss = fred[0] / 4194304.0f;
  __syncthreads();

  // counts: total + used
  int csum = 0, used = 0;
  for (int k = t; k < K_CODES; k += 256) { int c = counts[k]; csum += c; used += (c > 0); }
  ired[t] = csum; ired2[t] = used;
  __syncthreads();
  for (int k = 128; k > 0; k >>= 1) {
    if (t < k) { ired[t] += ired[t + k]; ired2[t] += ired2[t + k]; }
    __syncthreads();
  }
  float total = (float)ired[0];
  if (total < 1.f) total = 1.f;

  // entropy -> perplexity (p=0 terms contribute 0 in reference)
  float ent = 0.f;
  for (int k = t; k < K_CODES; k += 256) {
    int c = counts[k];
    if (c > 0) { float p = (float)c / total; ent += p * __logf(p); }
  }
  __syncthreads();
  fred[t] = ent;
  __syncthreads();
  for (int k = 128; k > 0; k >>= 1) { if (t < k) fred[t] += fred[t + k]; __syncthreads(); }

  if (t == 0) {
    float perp  = __expf(-fred[0]);
    float usedf = (float)ired2[0];
    out[OUT_SCAL + 0] = loss;                    // codebook_loss
    out[OUT_SCAL + 1] = loss;                    // commitment_loss (same value)
    out[OUT_SCAL + 2] = perp;                    // perplexity
    out[OUT_SCAL + 3] = usedf;                   // used_code_count
    out[OUT_SCAL + 4] = usedf / 8192.0f;         // used_code_fraction
    out[OUT_SCAL + 5] = 1.0f - usedf / 8192.0f;  // dead_code_fraction
  }
}

// ---------------------------------------------------------------------------
extern "C" void kernel_launch(void* const* d_in, const int* in_sizes, int n_in,
                              void* d_out, int out_size, void* d_ws, size_t ws_size,
                              hipStream_t stream) {
  const float* z = (const float*)d_in[0];   // z_e [16,256,32,32]
  const float* w = (const float*)d_in[1];   // embed [8192,256]
  float* out = (float*)d_out;

  char* ws = (char*)d_ws;
  unsigned short* ebf   = (unsigned short*)(ws + 0);         //  4 MB
  unsigned short* zbf   = (unsigned short*)(ws + 4194304);   //  8 MB
  float*          enorm = (float*)(ws + 12582912);           // 32 KB
  int*            cnts  = (int*)(ws + 12615680);             // 32 KB
  float*          part  = (float*)(ws + 12648448);           // 512 B (pad to 4K)
  float*          wdist = (float*)(ws + 12652544);           // 256 KB
  int*            widx  = (int*)(ws + 12914688);             // 256 KB

  vq_zero      <<<K_CODES / 256, 256, 0, stream>>>(cnts);
  vq_prep_embed<<<K_CODES,       256, 0, stream>>>(w, ebf, enorm);
  vq_prep_z    <<<N_TOT * D_DIM / 256, 256, 0, stream>>>(z, zbf);
  vq_main      <<<dim3(N_TOT / ROWS_PB, SPLITK), 256, 0, stream>>>(
                   zbf, ebf, enorm, wdist, widx);
  vq_combine   <<<N_TOT / ROWS_PB, 256, 0, stream>>>(z, w, wdist, widx,
                                                     out, cnts, part);
  vq_final     <<<1, 256, 0, stream>>>(part, cnts, out);
}
